// PublicLSTMQNetwork_30975304139466
// MI455X (gfx1250) — compile-verified, matrix operand order checked by reading, not run on hardware
//
#include <hip/hip_runtime.h>
#include <hip/hip_bf16.h>

// ---------------------------------------------------------------------------
// PublicLSTMQNetwork forward for MI455X (gfx1250, wave32, WMMA bf16->f32).
// T=128, B=256, OBS=1024 (512 priv + 512 publ), HID=512, ACT=21.
// All GEMMs run on v_wmma_f32_16x16x32_bf16 (bf16 inputs, f32 accumulate).
// Round 2: 16x64 wave tile in the batched GEMMs (4 accumulators, A-fragment
// reuse 4x) + global_prefetch on the packed-B stream.
// ---------------------------------------------------------------------------

#define T_STEPS 128
#define BATCH   256
#define HID     512
#define GATES   2048          // 4*HID
#define ROWS    32768         // T*B
#define ACT_N   21

typedef __attribute__((ext_vector_type(16))) __bf16 v16bf;
typedef __attribute__((ext_vector_type(8)))  float  v8f;

union Frag { uint4 u[2]; v16bf v; };

__device__ __forceinline__ unsigned short f2bf(float f) {
    unsigned u = __float_as_uint(f);
    unsigned r = ((u >> 16) & 1u) + 0x7FFFu;       // round-to-nearest-even
    return (unsigned short)((u + r) >> 16);
}
__device__ __forceinline__ float bf2f(unsigned short h) {
    return __uint_as_float(((unsigned)h) << 16);
}
__device__ __forceinline__ float sigm(float x) { return 1.0f / (1.0f + __expf(-x)); }

// ---------------------------------------------------------------------------
// Weight packing: W is KxN row-major f32. Packed layout (bf16):
//   dst[(((kt*ntiles + nt)*32 + lane)*16 + j]
// lane holds column nt*16 + lane%16, K values kt*32 + (lane>=16?8:0) +
// (j<8 ? j : j+8)  -> mirrors the 16-bit WMMA fragment layout so the GEMM
// waves load B with two coalesced b128 reads per lane.
// ---------------------------------------------------------------------------
__global__ void pack_b_kernel(const float* __restrict__ W,
                              unsigned short* __restrict__ dst, int K, int N) {
    size_t idx = (size_t)blockIdx.x * blockDim.x + threadIdx.x;
    size_t total = (size_t)K * N;
    if (idx >= total) return;
    int ntiles = N >> 4;
    int j    = (int)(idx & 15);
    int lane = (int)((idx >> 4) & 31);
    int nt   = (int)((idx >> 9) % ntiles);
    int kt   = (int)(idx / ((size_t)ntiles << 9));
    int k    = kt * 32 + ((lane >= 16) ? 8 : 0) + (j < 8 ? j : j + 8);
    int col  = nt * 16 + (lane & 15);
    dst[idx] = f2bf(W[(size_t)k * N + col]);
}

// Build padded head weight [Wv | Wa | 0...] : 512 x 32 f32.
__global__ void merge_head_kernel(const float* __restrict__ Wv,
                                  const float* __restrict__ Wa,
                                  float* __restrict__ Whead) {
    int idx = blockIdx.x * blockDim.x + threadIdx.x;
    if (idx >= HID * 32) return;
    int k = idx >> 5, c = idx & 31;
    float v = 0.0f;
    if (c == 0)          v = Wv[k];
    else if (c <= ACT_N) v = Wa[(size_t)k * ACT_N + (c - 1)];
    Whead[idx] = v;
}

// Split obs (ROWS x 1024 f32) into priv / publ bf16 activations (ROWS x 512).
__global__ void split_obs_kernel(const float* __restrict__ obs,
                                 unsigned short* __restrict__ privb,
                                 unsigned short* __restrict__ publb) {
    size_t idx = (size_t)blockIdx.x * blockDim.x + threadIdx.x;
    if (idx >= (size_t)ROWS * 1024) return;
    size_t row = idx >> 10;
    int    c   = (int)(idx & 1023);
    unsigned short b = f2bf(obs[idx]);
    if (c < 512) privb[row * 512 + c]         = b;
    else         publb[row * 512 + (c - 512)] = b;
}

__global__ void init_state_kernel(const float* __restrict__ c0,
                                  const float* __restrict__ h0,
                                  float* __restrict__ c_st,
                                  unsigned short* __restrict__ h_bf) {
    int idx = blockIdx.x * blockDim.x + threadIdx.x;
    if (idx >= BATCH * HID) return;
    c_st[idx] = c0[idx];
    h_bf[idx] = f2bf(h0[idx]);
}

__global__ void mul_o_kernel(const unsigned short* __restrict__ a,
                             const unsigned short* __restrict__ b,
                             unsigned short* __restrict__ o, size_t n) {
    size_t idx = (size_t)blockIdx.x * blockDim.x + threadIdx.x;
    if (idx >= n) return;
    o[idx] = f2bf(bf2f(a[idx]) * bf2f(b[idx]));
}

// ---------------------------------------------------------------------------
// Generic GEMM: C(MxN bf16) = [relu](A(MxK bf16 row-major) @ Bpacked + bias).
// One wave32 per 16x64 tile (4 f32 accumulators, A fragment reused 4x),
// 4 waves per block, K stepped 32 at a time through v_wmma_f32_16x16x32_bf16.
// Per K-chunk: 2 A b128 loads + 8 B b128 loads -> 4 WMMA. LDS repack for
// coalesced 16B stores. Requires N % 64 == 0.
// ---------------------------------------------------------------------------
__global__ void __launch_bounds__(128)
gemm_bf16_kernel(const unsigned short* __restrict__ A,
                 const unsigned short* __restrict__ Bp,
                 const float* __restrict__ bias,
                 unsigned short* __restrict__ C,
                 int N, int K, int relu) {
    __shared__ __align__(16) unsigned short smem[4][16][64];
    int lane = threadIdx.x & 31;
    int w    = threadIdx.x >> 5;
    int ntiles = N >> 4;
    int nblk   = N >> 6;                 // 64-col blocks
    int tile = blockIdx.x * 4 + w;
    int mt = tile / nblk, nb = tile % nblk;
    int m0 = mt << 4, n0 = nb << 6;
    int nt0 = nb * 4;

    v8f acc[4];
#pragma unroll
    for (int j = 0; j < 4; ++j) {
        float b = bias ? bias[n0 + j * 16 + (lane & 15)] : 0.0f;
#pragma unroll
        for (int r = 0; r < 8; ++r) acc[j][r] = b;
    }

    const uint4* arow = (const uint4*)(A + (size_t)(m0 + (lane & 15)) * K);
    int asel = (lane < 16) ? 0 : 1;
    int nkt = K >> 5;
    for (int kk = 0; kk < nkt; ++kk) {
        Frag a;
        a.u[0] = arow[4 * kk + asel];
        a.u[1] = arow[4 * kk + 2 + asel];
        // Prefetch the B stream two K-chunks ahead (global_prefetch_b8).
        __builtin_prefetch(Bp + ((((size_t)(kk + 2) * ntiles + nt0) * 32 + lane) << 4), 0, 3);
#pragma unroll
        for (int j = 0; j < 4; ++j) {
            Frag bf;
            const uint4* bp =
                (const uint4*)(Bp + ((((size_t)kk * ntiles + nt0 + j) * 32 + lane) << 4));
            bf.u[0] = bp[0];
            bf.u[1] = bp[1];
            acc[j] = __builtin_amdgcn_wmma_f32_16x16x32_bf16(false, a.v, false, bf.v,
                                                             (short)0, acc[j], false, false);
        }
    }
    int rbase = (lane < 16) ? 0 : 8;
#pragma unroll
    for (int j = 0; j < 4; ++j)
#pragma unroll
        for (int r = 0; r < 8; ++r) {
            float x = acc[j][r];
            if (relu) x = x > 0.0f ? x : 0.0f;
            smem[w][rbase + r][j * 16 + (lane & 15)] = f2bf(x);
        }
    __syncthreads();
    // 16 rows x 128B per row = 128 uint4 chunks; 32 lanes x 4 iterations.
#pragma unroll
    for (int i = 0; i < 4; ++i) {
        int idx = i * 32 + lane;
        int row = idx >> 3, ch = idx & 7;
        uint4 v = *(const uint4*)&smem[w][row][ch * 8];
        *(uint4*)(C + (size_t)(m0 + row) * N + n0 + ch * 8) = v;
    }
}

// ---------------------------------------------------------------------------
// One LSTM step. Grid (16 mtiles, 32 ntiles), block = 128 = 4 waves.
// Wave g computes the 16x16 gate-g tile: acc = xg[t] + b_lstm + h @ Wh.
// dones mask fused into the h fragment load and c read. h double-buffered.
// ---------------------------------------------------------------------------
__global__ void __launch_bounds__(128)
lstm_step_kernel(const unsigned short* __restrict__ xg,
                 const unsigned short* __restrict__ Whp,
                 const float* __restrict__ b_lstm,
                 const unsigned char* __restrict__ dones,
                 const unsigned short* __restrict__ h_in,
                 unsigned short* __restrict__ h_out,
                 float* __restrict__ c_st,
                 unsigned short* __restrict__ publ_o,
                 float* __restrict__ cT_out,
                 float* __restrict__ hT_out, int t) {
    __shared__ __align__(16) float gl[4][16][16];
    int lane = threadIdx.x & 31;
    int g    = threadIdx.x >> 5;
    int m0 = blockIdx.x << 4;
    int n0 = blockIdx.y << 4;
    int rbase = (lane < 16) ? 0 : 8;
    int colg = g * HID + n0 + (lane & 15);      // column in [0,2048)

    v8f acc;
    float bl = b_lstm[colg];
    const unsigned short* xgt = xg + (size_t)t * BATCH * GATES;
#pragma unroll
    for (int r = 0; r < 8; ++r)
        acc[r] = bl + bf2f(xgt[(size_t)(m0 + rbase + r) * GATES + colg]);

    int arowi = m0 + (lane & 15);
    int dn = dones[t * BATCH + arowi] ? 1 : 0;
    const uint4* arow = (const uint4*)(h_in + (size_t)arowi * HID);
    int asel = (lane < 16) ? 0 : 1;
    int ntg = g * 32 + blockIdx.y;              // Wh has 128 n-tiles
    for (int kk = 0; kk < 16; ++kk) {
        Frag a, bf;
        if (dn) {
            a.u[0] = make_uint4(0u, 0u, 0u, 0u);
            a.u[1] = a.u[0];
        } else {
            a.u[0] = arow[4 * kk + asel];
            a.u[1] = arow[4 * kk + 2 + asel];
        }
        const uint4* bp = (const uint4*)(Whp + ((((size_t)kk * 128 + ntg) * 32 + lane) << 4));
        bf.u[0] = bp[0];
        bf.u[1] = bp[1];
        acc = __builtin_amdgcn_wmma_f32_16x16x32_bf16(false, a.v, false, bf.v,
                                                      (short)0, acc, false, false);
    }
#pragma unroll
    for (int r = 0; r < 8; ++r) gl[g][rbase + r][lane & 15] = acc[r];
    __syncthreads();

    for (int e = threadIdx.x; e < 256; e += 128) {
        int m = e >> 4, n = e & 15;
        int brow = m0 + m;
        int col  = n0 + n;
        float iv = sigm(gl[0][m][n]);
        float fv = sigm(gl[1][m][n]);
        float gv = tanhf(gl[2][m][n]);
        float ov = sigm(gl[3][m][n]);
        float cold = dones[t * BATCH + brow] ? 0.0f : c_st[(size_t)brow * HID + col];
        float cnew = fv * cold + iv * gv;
        float hnew = ov * tanhf(cnew);
        c_st[(size_t)brow * HID + col] = cnew;
        unsigned short hb = f2bf(hnew);
        h_out[(size_t)brow * HID + col] = hb;
        publ_o[((size_t)t * BATCH + brow) * HID + col] = hb;
        if (t == T_STEPS - 1) {
            cT_out[(size_t)brow * HID + col] = cnew;
            hT_out[(size_t)brow * HID + col] = hnew;
        }
    }
}

// ---------------------------------------------------------------------------
// Heads: out = o @ [Wv|Wa]pad + [bv|ba], then q = v + a. One wave per 16-row
// chunk computes both 16x16 tiles (cols 0-31), 4 waves per block.
// ---------------------------------------------------------------------------
__global__ void __launch_bounds__(128)
heads_kernel(const unsigned short* __restrict__ O,
             const unsigned short* __restrict__ Whp,
             const float* __restrict__ bv,
             const float* __restrict__ ba,
             float* __restrict__ q) {
    __shared__ __align__(16) float hl[4][16][32];
    int lane = threadIdx.x & 31;
    int w    = threadIdx.x >> 5;
    int m0 = (blockIdx.x * 4 + w) << 4;
    int coln = lane & 15;

    float bias0 = (coln == 0) ? bv[0] : ba[coln - 1];
    int c1 = 16 + coln;
    float bias1 = (c1 <= ACT_N) ? ba[c1 - 1] : 0.0f;
    v8f a0, a1;
#pragma unroll
    for (int r = 0; r < 8; ++r) { a0[r] = bias0; a1[r] = bias1; }

    const uint4* arow = (const uint4*)(O + (size_t)(m0 + coln) * HID);
    int asel = (lane < 16) ? 0 : 1;
    for (int kk = 0; kk < 16; ++kk) {
        Frag a, b0, b1;
        a.u[0] = arow[4 * kk + asel];
        a.u[1] = arow[4 * kk + 2 + asel];
        const uint4* p0 = (const uint4*)(Whp + ((((size_t)kk * 2 + 0) * 32 + lane) << 4));
        const uint4* p1 = (const uint4*)(Whp + ((((size_t)kk * 2 + 1) * 32 + lane) << 4));
        b0.u[0] = p0[0]; b0.u[1] = p0[1];
        b1.u[0] = p1[0]; b1.u[1] = p1[1];
        a0 = __builtin_amdgcn_wmma_f32_16x16x32_bf16(false, a.v, false, b0.v,
                                                     (short)0, a0, false, false);
        a1 = __builtin_amdgcn_wmma_f32_16x16x32_bf16(false, a.v, false, b1.v,
                                                     (short)0, a1, false, false);
    }
    int rbase = (lane < 16) ? 0 : 8;
#pragma unroll
    for (int r = 0; r < 8; ++r) {
        hl[w][rbase + r][coln]      = a0[r];
        hl[w][rbase + r][16 + coln] = a1[r];
    }
    __syncthreads();

    int row = lane & 15;
    int jbase  = (lane < 16) ? 0 : 11;
    int jcount = (lane < 16) ? 11 : 10;
    float v = hl[w][row][0];
    size_t rg = (size_t)(m0 + row);
    for (int j = 0; j < jcount; ++j)
        q[rg * ACT_N + jbase + j] = v + hl[w][row][1 + jbase + j];
}

// ---------------------------------------------------------------------------
// Host side
// ---------------------------------------------------------------------------
extern "C" void kernel_launch(void* const* d_in, const int* in_sizes, int n_in,
                              void* d_out, int out_size, void* d_ws, size_t ws_size,
                              hipStream_t stream) {
    const float* c0   = (const float*)d_in[0];
    const float* h0   = (const float*)d_in[1];
    const float* obs  = (const float*)d_in[2];
    const unsigned char* dones = (const unsigned char*)d_in[3];
    const float* Wp1  = (const float*)d_in[4];
    const float* bp1  = (const float*)d_in[5];
    const float* Wp2  = (const float*)d_in[6];
    const float* bp2  = (const float*)d_in[7];
    const float* Wp3  = (const float*)d_in[8];
    const float* bp3  = (const float*)d_in[9];
    const float* Wpub = (const float*)d_in[10];
    const float* bpub = (const float*)d_in[11];
    const float* Wi   = (const float*)d_in[12];
    const float* Wh   = (const float*)d_in[13];
    const float* bl   = (const float*)d_in[14];
    const float* Wv   = (const float*)d_in[15];
    const float* bv   = (const float*)d_in[16];
    const float* Wa   = (const float*)d_in[17];
    const float* ba   = (const float*)d_in[18];
    (void)in_sizes; (void)n_in; (void)out_size; (void)ws_size;

    float* out = (float*)d_out;
    float* cT  = out;                   // 256*512
    float* hT  = out + BATCH * HID;     // 256*512
    float* q   = out + 2 * BATCH * HID; // 128*256*21

    // Workspace carve-out (all offsets 256B aligned).
    size_t off = 0;
    char* ws = (char*)d_ws;
    auto carve = [&](size_t bytes) -> char* {
        char* p = ws + off;
        off += (bytes + 255) & ~(size_t)255;
        return p;
    };
    unsigned short* wp1p   = (unsigned short*)carve((size_t)HID * HID * 2);
    unsigned short* wp2p   = (unsigned short*)carve((size_t)HID * HID * 2);
    unsigned short* wp3p   = (unsigned short*)carve((size_t)HID * HID * 2);
    unsigned short* wpubp  = (unsigned short*)carve((size_t)HID * HID * 2);
    unsigned short* wip    = (unsigned short*)carve((size_t)HID * GATES * 2);
    unsigned short* whp    = (unsigned short*)carve((size_t)HID * GATES * 2);
    unsigned short* wheadp = (unsigned short*)carve((size_t)HID * 32 * 2);
    float*          wheadf = (float*)carve((size_t)HID * 32 * 4);
    unsigned short* privin = (unsigned short*)carve((size_t)ROWS * HID * 2);
    unsigned short* publin = (unsigned short*)carve((size_t)ROWS * HID * 2);
    unsigned short* actA   = (unsigned short*)carve((size_t)ROWS * HID * 2);
    unsigned short* actB   = (unsigned short*)carve((size_t)ROWS * HID * 2);
    unsigned short* xgb    = (unsigned short*)carve((size_t)ROWS * GATES * 2);
    unsigned short* publo  = (unsigned short*)carve((size_t)ROWS * HID * 2);
    float*          cst    = (float*)carve((size_t)BATCH * HID * 4);
    unsigned short* hbf0   = (unsigned short*)carve((size_t)BATCH * HID * 2);
    unsigned short* hbf1   = (unsigned short*)carve((size_t)BATCH * HID * 2);

    // 1) Weight prep.
    merge_head_kernel<<<(HID * 32 + 255) / 256, 256, 0, stream>>>(Wv, Wa, wheadf);
    pack_b_kernel<<<(HID * HID) / 256, 256, 0, stream>>>(Wp1, wp1p, HID, HID);
    pack_b_kernel<<<(HID * HID) / 256, 256, 0, stream>>>(Wp2, wp2p, HID, HID);
    pack_b_kernel<<<(HID * HID) / 256, 256, 0, stream>>>(Wp3, wp3p, HID, HID);
    pack_b_kernel<<<(HID * HID) / 256, 256, 0, stream>>>(Wpub, wpubp, HID, HID);
    pack_b_kernel<<<(HID * GATES) / 256, 256, 0, stream>>>(Wi, wip, HID, GATES);
    pack_b_kernel<<<(HID * GATES) / 256, 256, 0, stream>>>(Wh, whp, HID, GATES);
    pack_b_kernel<<<(HID * 32) / 256, 256, 0, stream>>>(wheadf, wheadp, HID, 32);

    // 2) Observation split + bf16 convert.
    split_obs_kernel<<<((size_t)ROWS * 1024) / 256, 256, 0, stream>>>(obs, privin, publin);

    // 3) Batched encoder GEMMs (wave-per-16x64-tile WMMA).
    int blocks512 = (ROWS / 16) * (HID / 64) / 4;    // 4096 blocks
    gemm_bf16_kernel<<<blocks512, 128, 0, stream>>>(privin, wp1p, bp1, actA, HID, HID, 1);
    gemm_bf16_kernel<<<blocks512, 128, 0, stream>>>(actA, wp2p, bp2, actB, HID, HID, 1);
    gemm_bf16_kernel<<<blocks512, 128, 0, stream>>>(actB, wp3p, bp3, actA, HID, HID, 1);  // priv_o
    gemm_bf16_kernel<<<blocks512, 128, 0, stream>>>(publin, wpubp, bpub, actB, HID, HID, 1); // publ_x
    int blocksXG = (ROWS / 16) * (GATES / 64) / 4;   // 16384 blocks
    gemm_bf16_kernel<<<blocksXG, 128, 0, stream>>>(actB, wip, nullptr, xgb, GATES, HID, 0); // x_gates

    // 4) LSTM scan: 128 sequential step kernels, h double-buffered in bf16.
    init_state_kernel<<<(BATCH * HID) / 256, 256, 0, stream>>>(c0, h0, cst, hbf0);
    for (int t = 0; t < T_STEPS; ++t) {
        unsigned short* hin  = (t & 1) ? hbf1 : hbf0;
        unsigned short* hout = (t & 1) ? hbf0 : hbf1;
        lstm_step_kernel<<<dim3(BATCH / 16, HID / 16), 128, 0, stream>>>(
            xgb, whp, bl, dones, hin, hout, cst, publo, cT, hT, t);
    }

    // 5) o = priv_o * publ_o, then fused value/advantage heads.
    mul_o_kernel<<<((size_t)ROWS * HID) / 256, 256, 0, stream>>>(actA, publo, actB,
                                                                 (size_t)ROWS * HID);
    heads_kernel<<<ROWS / 64, 128, 0, stream>>>(actB, wheadp, bv, ba, q);
}
